// FairADG_6296422056688
// MI455X (gfx1250) — compile-verified, matrix operand order checked by reading, not run on hardware
//
#include <hip/hip_runtime.h>
#include <math.h>

#define N_NODES 50000
#define N_EDGES 800000
#define NFEAT   128
#define HIDDEN  256
#define CHANNELS 4
#define PCD     64

typedef __attribute__((ext_vector_type(16))) __bf16          v16bf;
typedef __attribute__((ext_vector_type(8)))  float           v8f;
typedef __attribute__((ext_vector_type(16))) unsigned short  v16u;

union BfFrag {
    v16u           u;
    v16bf          b;
    unsigned short s[16];
    uint4          q[2];
};

__device__ __forceinline__ unsigned short f2bf(float f) {
    unsigned int u = __float_as_uint(f);
    u += 0x7FFFu + ((u >> 16) & 1u);      // round-to-nearest-even
    return (unsigned short)(u >> 16);
}

// A/B fragment for v_wmma_f32_16x16x32_bf16, 16-bit operand layout:
// lanes 0-15:  K = lo..lo+7 in regs 0..3, K = 16+lo..16+lo+7 in regs 4..7 (lo=0)
// lanes 16-31: same row, lo=8.  Caller passes p already offset by lo.
__device__ __forceinline__ v16bf make_frag_bf(const unsigned short* __restrict__ p) {
    BfFrag r;
    r.q[0] = reinterpret_cast<const uint4*>(p)[0];
    r.q[1] = reinterpret_cast<const uint4*>(p + 16)[0];
    return r.b;
}

// ---- x f32 -> bf16, one pass (each element used 16x by GEMM1; convert once)
__global__ void cvtx_kernel(const float* __restrict__ x, unsigned short* __restrict__ xbf) {
    const size_t n = (size_t)N_NODES * NFEAT / 4;       // process float4 chunks
    size_t i = (size_t)blockIdx.x * blockDim.x + threadIdx.x;
    const size_t stride = (size_t)gridDim.x * blockDim.x;
    for (; i < n; i += stride) {
        const float4 f = reinterpret_cast<const float4*>(x)[i];
        ushort4 o;
        o.x = f2bf(f.x); o.y = f2bf(f.y); o.z = f2bf(f.z); o.w = f2bf(f.w);
        reinterpret_cast<ushort4*>(xbf)[i] = o;
    }
}

// ---- weight packing: lin_W [4,64,128] -> bf16 (already N-major/K-contig);
//      conv_W [k][d][e] -> cwt [k][e][d] bf16 (so B frag reads K contiguously)
__global__ void pack_kernel(const float* __restrict__ linW, const float* __restrict__ convW,
                            unsigned short* __restrict__ wlin, unsigned short* __restrict__ cwt) {
    const int i = blockIdx.x * blockDim.x + threadIdx.x;
    if (i < CHANNELS * PCD * NFEAT) {                       // 32768
        wlin[i] = f2bf(linW[i]);
    } else if (i < CHANNELS * PCD * NFEAT + CHANNELS * PCD * PCD) {
        const int j = i - CHANNELS * PCD * NFEAT;
        const int k = j >> 12;
        const int e = (j >> 6) & 63;
        const int d = j & 63;
        cwt[j] = f2bf(convW[(k << 12) | (d << 6) | e]);
    }
}

__global__ void zero_kernel(float* __restrict__ p, size_t n) {
    size_t i = (size_t)blockIdx.x * blockDim.x + threadIdx.x;
    const size_t stride = (size_t)gridDim.x * blockDim.x;
    for (; i < n; i += stride) p[i] = 0.0f;
}

// ---- per-node assigner projections: pa = x@W1a^T, pb = x@W1b^T   (wave per node)
__global__ void nodeproj_kernel(const float* __restrict__ x, const float* __restrict__ asgW1,
                                float* __restrict__ pa, float* __restrict__ pb) {
    const int wid  = (int)((blockIdx.x * blockDim.x + threadIdx.x) >> 5);
    if (wid >= N_NODES) return;
    const int lane = threadIdx.x & 31;
    const float* xr = x + (size_t)wid * NFEAT;
    const float xv0 = xr[lane], xv1 = xr[lane + 32], xv2 = xr[lane + 64], xv3 = xr[lane + 96];
    float acc[8];
#pragma unroll
    for (int c = 0; c < 4; ++c) {
        const float* wr = asgW1 + c * (2 * NFEAT);
        acc[c]     = xv0 * wr[lane]          + xv1 * wr[lane + 32]
                   + xv2 * wr[lane + 64]     + xv3 * wr[lane + 96];
        acc[4 + c] = xv0 * wr[NFEAT + lane]      + xv1 * wr[NFEAT + lane + 32]
                   + xv2 * wr[NFEAT + lane + 64] + xv3 * wr[NFEAT + lane + 96];
    }
#pragma unroll
    for (int off = 16; off > 0; off >>= 1)
#pragma unroll
        for (int i = 0; i < 8; ++i) acc[i] += __shfl_xor(acc[i], off, 32);
    if (lane == 0) {
#pragma unroll
        for (int c = 0; c < 4; ++c) {
            pa[wid * 4 + c] = acc[c];
            pb[wid * 4 + c] = acc[4 + c];
        }
    }
}

// ---- GEMM1: z[N,256] = x[N,128] @ WlinT^T + lin_b  -> bf16   (wave per 16x16 tile)
__global__ void gemm1_kernel(const unsigned short* __restrict__ xbf,
                             const unsigned short* __restrict__ wlin,
                             const float* __restrict__ linb, unsigned short* __restrict__ zbf) {
    const int wave = (int)((blockIdx.x * blockDim.x + threadIdx.x) >> 5);
    const int lane = threadIdx.x & 31;
    const int mtile = wave >> 4;       // 3125 tiles over rows
    const int ntile = wave & 15;       // 16 tiles over 256 cols
    const int lm = lane & 15;
    const int hi = lane >> 4;
    const int lo = hi << 3;
    const unsigned short* arow = xbf  + (size_t)(mtile * 16 + lm) * NFEAT;
    const unsigned short* brow = wlin + (size_t)(ntile * 16 + lm) * NFEAT;
    v8f acc = {};
#pragma unroll
    for (int kb = 0; kb < 4; ++kb) {
        v16bf a = make_frag_bf(arow + kb * 32 + lo);
        v16bf b = make_frag_bf(brow + kb * 32 + lo);
        acc = __builtin_amdgcn_wmma_f32_16x16x32_bf16(false, a, false, b, (short)0, acc, false, false);
    }
    const int ncol  = ntile * 16 + lm;
    const float bias = linb[ncol];
    const int mbase = mtile * 16 + hi * 8;
#pragma unroll
    for (int j = 0; j < 8; ++j)
        zbf[(size_t)(mbase + j) * HIDDEN + ncol] = f2bf(acc[j] + bias);
}

// ---- GEMM2: per channel k, c[:,k*64+e] = z[:,k*64+:] @ conv_W[k]  -> f32
__global__ void gemm2_kernel(const unsigned short* __restrict__ zbf,
                             const unsigned short* __restrict__ cwt,
                             float* __restrict__ cfeat) {
    const int wave = (int)((blockIdx.x * blockDim.x + threadIdx.x) >> 5);
    const int lane = threadIdx.x & 31;
    const int mtile = wave >> 4;
    const int ntile = wave & 15;
    const int ch   = ntile >> 2;
    const int nsub = (ntile & 3) * 16;
    const int lm = lane & 15;
    const int hi = lane >> 4;
    const int lo = hi << 3;
    const unsigned short* arow = zbf + (size_t)(mtile * 16 + lm) * HIDDEN + ch * PCD;
    const unsigned short* brow = cwt + (size_t)(ch * PCD + nsub + lm) * PCD;
    v8f acc = {};
#pragma unroll
    for (int kb = 0; kb < 2; ++kb) {
        v16bf a = make_frag_bf(arow + kb * 32 + lo);
        v16bf b = make_frag_bf(brow + kb * 32 + lo);
        acc = __builtin_amdgcn_wmma_f32_16x16x32_bf16(false, a, false, b, (short)0, acc, false, false);
    }
    const int ncol  = ch * PCD + nsub + lm;
    const int mbase = mtile * 16 + hi * 8;
#pragma unroll
    for (int j = 0; j < 8; ++j)
        cfeat[(size_t)(mbase + j) * HIDDEN + ncol] = acc[j];
}

// ---- edge phase: wave per edge; assigner softmax + weighted scatter-add
__global__ void edge_kernel(const int* __restrict__ erow, const int* __restrict__ ecol,
                            const float* __restrict__ pa, const float* __restrict__ pb,
                            const float* __restrict__ cfeat,
                            const float* __restrict__ b1, const float* __restrict__ W2,
                            const float* __restrict__ b2, float* __restrict__ agg) {
    const int wid = (int)((blockIdx.x * blockDim.x + threadIdx.x) >> 5);
    if (wid >= N_EDGES) return;
    const int lane = threadIdx.x & 31;
    const int col = ecol[wid];
    const int row = erow[wid];
    const float4 a4 = reinterpret_cast<const float4*>(pa)[col];
    const float4 b4 = reinterpret_cast<const float4*>(pb)[row];
    const float t0 = a4.x + b4.x + b1[0];
    const float t1 = a4.y + b4.y + b1[1];
    const float t2 = a4.z + b4.z + b1[2];
    const float t3 = a4.w + b4.w + b1[3];
    float s[4];
#pragma unroll
    for (int i = 0; i < 4; ++i)
        s[i] = b2[i] + W2[i*4+0]*t0 + W2[i*4+1]*t1 + W2[i*4+2]*t2 + W2[i*4+3]*t3;
    const float m  = fmaxf(fmaxf(s[0], s[1]), fmaxf(s[2], s[3]));
    const float e0 = __expf(s[0]-m), e1 = __expf(s[1]-m), e2 = __expf(s[2]-m), e3 = __expf(s[3]-m);
    const float inv = 1.0f / (e0 + e1 + e2 + e3);
    const float w[4] = { e0*inv, e1*inv, e2*inv, e3*inv };
    const float wk = w[lane >> 3];                 // 8 lanes per channel, 8 elems/lane
    const float4* cp = reinterpret_cast<const float4*>(cfeat + (size_t)col * HIDDEN + lane * 8);
    const float4 c0 = cp[0], c1 = cp[1];
    float* ap = agg + (size_t)row * HIDDEN + lane * 8;
    atomicAdd(ap + 0, c0.x * wk); atomicAdd(ap + 1, c0.y * wk);
    atomicAdd(ap + 2, c0.z * wk); atomicAdd(ap + 3, c0.w * wk);
    atomicAdd(ap + 4, c1.x * wk); atomicAdd(ap + 5, c1.y * wk);
    atomicAdd(ap + 6, c1.z * wk); atomicAdd(ap + 7, c1.w * wk);
}

// ---- finalize: agg += bias, per-channel L2 normalize -> h (in place), classifier logits
__global__ void finalize_kernel(const float* __restrict__ chan_bias,
                                const float* __restrict__ clsW,
                                const float* __restrict__ clsb,
                                float* __restrict__ out) {
    const int node = blockIdx.x;
    const int t = threadIdx.x;                     // 0..255
    __shared__ float sh[HIDDEN];
    float* aggrow = out + (size_t)node * HIDDEN;
    const float v = aggrow[t] + chan_bias[t];
    sh[t] = v * v;
    __syncthreads();
#pragma unroll
    for (int off = 32; off > 0; off >>= 1) {       // 64-wide per-channel reduce
        if ((t & 63) < off) sh[t] += sh[t + off];
        __syncthreads();
    }
    const float norm = fmaxf(sqrtf(sh[t & ~63]), 1e-12f);
    const float h = v / norm;
    __syncthreads();
    aggrow[t] = h;
    sh[t] = h * clsW[t];
    __syncthreads();
#pragma unroll
    for (int off = 128; off > 0; off >>= 1) {
        if (t < off) sh[t] += sh[t + off];
        __syncthreads();
    }
    const float o0 = sh[0];
    __syncthreads();
    sh[t] = h * clsW[HIDDEN + t];
    __syncthreads();
#pragma unroll
    for (int off = 128; off > 0; off >>= 1) {
        if (t < off) sh[t] += sh[t + off];
        __syncthreads();
    }
    if (t == 0) {
        float* logits = out + (size_t)N_NODES * HIDDEN;
        logits[node * 2 + 0] = o0    + clsb[0];
        logits[node * 2 + 1] = sh[0] + clsb[1];
    }
}

extern "C" void kernel_launch(void* const* d_in, const int* in_sizes, int n_in,
                              void* d_out, int out_size, void* d_ws, size_t ws_size,
                              hipStream_t stream) {
    (void)in_sizes; (void)n_in; (void)out_size; (void)ws_size;
    const float* x     = (const float*)d_in[0];
    const int*   erow  = (const int*)  d_in[1];
    const int*   ecol  = (const int*)  d_in[2];
    const float* asgW1 = (const float*)d_in[3];
    const float* asgb1 = (const float*)d_in[4];
    const float* asgW2 = (const float*)d_in[5];
    const float* asgb2 = (const float*)d_in[6];
    const float* linW  = (const float*)d_in[7];
    const float* linb  = (const float*)d_in[8];
    const float* convW = (const float*)d_in[9];
    const float* chanb = (const float*)d_in[10];
    const float* clsW  = (const float*)d_in[11];
    const float* clsb  = (const float*)d_in[12];
    float* out = (float*)d_out;

    char* ws = (char*)d_ws;
    size_t off = 0;
    auto take = [&](size_t bytes) -> char* {
        char* p = ws + off;
        off = (off + bytes + 255) & ~(size_t)255;
        return p;
    };
    unsigned short* wlin  = (unsigned short*)take((size_t)CHANNELS * PCD * NFEAT * 2);
    unsigned short* cwt   = (unsigned short*)take((size_t)CHANNELS * PCD * PCD * 2);
    unsigned short* xbf   = (unsigned short*)take((size_t)N_NODES * NFEAT * 2);
    unsigned short* zbf   = (unsigned short*)take((size_t)N_NODES * HIDDEN * 2);
    float*          cfeat = (float*)         take((size_t)N_NODES * HIDDEN * 4);
    float*          pa    = (float*)         take((size_t)N_NODES * 4 * 4);
    float*          pb    = (float*)         take((size_t)N_NODES * 4 * 4);

    pack_kernel    <<<192, 256, 0, stream>>>(linW, convW, wlin, cwt);
    cvtx_kernel    <<<2048, 256, 0, stream>>>(x, xbf);
    zero_kernel    <<<2048, 256, 0, stream>>>(out, (size_t)N_NODES * HIDDEN);
    nodeproj_kernel<<<N_NODES / 8, 256, 0, stream>>>(x, asgW1, pa, pb);
    gemm1_kernel   <<<6250, 256, 0, stream>>>(xbf, wlin, linb, zbf);    // 50000 wave-tiles
    gemm2_kernel   <<<6250, 256, 0, stream>>>(zbf, cwt, cfeat);
    edge_kernel    <<<N_EDGES / 8, 256, 0, stream>>>(erow, ecol, pa, pb, cfeat,
                                                     asgb1, asgW2, asgb2, out);
    finalize_kernel<<<N_NODES, 256, 0, stream>>>(chanb, clsW, clsb, out);
}
